// SelfAttention_16011638079744
// MI455X (gfx1250) — matrix operand order repaired and checked by reading, not compiled, not run on hardware
//
#include <hip/hip_runtime.h>
#include <hip/hip_bf16.h>

// Channel self-attention, B=8 C=512 H=W=64 (N=4096), fp32 in/out.
// Pipeline (all bf16 WMMA 16x16x32, f32 accumulate):
//   1) qkv_gemm  : Q/K/V[b][c][n] = bf16(W x X + bias)    (X transposed in LDS)
//   2) qk_gemm   : S[b][i][j] = sum_n Q[i][n] K[j][n]     (no transposes needed)
//   3) softmax   : P[b][i][j] = softmax_j(S)              (bf16 out)
//   4) av_gemm   : out[b][i][n] = sum_j P[i][j] V[j][n]   (V transposed in LDS)
// Intermediates (~110 MB) fit in the 192 MB L2, so stages 2-4 run from L2.

typedef __attribute__((ext_vector_type(16))) __bf16 v16bf;
typedef __attribute__((ext_vector_type(8)))  __bf16 v8bf;
typedef __attribute__((ext_vector_type(8)))  float  v8f;

union ABFrag { v16bf v; v8bf h[2]; };

#define BDIM 8
#define CDIM 512
#define NDIM 4096
#define LPITCH 40   // LDS row pitch in bf16 (80B: 16B-aligned, bank-friendly)

// A fragment: row-major K-contiguous LDS tile, tile origin (m0 rows, k=0..31)
__device__ inline v16bf load_a_frag(const __bf16* lA, int m0, int lane) {
    ABFrag f;
    const int row = m0 + (lane & 15);
    const int kb  = (lane >> 4) * 8;            // lanes 16-31: K base 8 / 24
    const __bf16* p = lA + row * LPITCH + kb;
    f.h[0] = *(const v8bf*)(p);                 // K = kb .. kb+7
    f.h[1] = *(const v8bf*)(p + 16);            // K = kb+16 .. kb+23
    return f.v;
}

// B fragment: LDS tile holds B^T as [n][k] (k contiguous), tile origin n0
__device__ inline v16bf load_b_frag(const __bf16* lB, int n0, int lane) {
    ABFrag f;
    const int col = n0 + (lane & 15);
    const int kb  = (lane >> 4) * 8;            // lanes 16-31: K base 8 / 24
    const __bf16* p = lB + col * LPITCH + kb;
    f.h[0] = *(const v8bf*)(p);
    f.h[1] = *(const v8bf*)(p + 16);
    return f.v;
}

// ---------------- Stage 1: Y = bf16(W * X + bias), tile 128(M) x 256(N), K-step 32
__global__ __launch_bounds__(256) void qkv_gemm(
    const float* __restrict__ W, const float* __restrict__ bias,
    const float* __restrict__ X, __bf16* __restrict__ Y)
{
    __shared__ __bf16 lA[128 * LPITCH];
    __shared__ __bf16 lB[256 * LPITCH];
    __shared__ float  lbias[128];

    const int b   = blockIdx.z;
    const int m0b = blockIdx.y * 128;
    const int n0b = blockIdx.x * 256;
    const float*  Xb = X + (size_t)b * CDIM * NDIM;
    __bf16*       Yb = Y + (size_t)b * CDIM * NDIM;

    const int tid  = threadIdx.x;
    if (tid < 128) lbias[tid] = bias[m0b + tid];
    const int lane = tid & 31;
    const int wave = tid >> 5;
    const int wm = (wave & 1) * 64;             // 2 x 4 wave grid, 64x64 per wave
    const int wn = (wave >> 1) * 64;

    v8f acc[4][4] = {};
    for (int k0 = 0; k0 < CDIM; k0 += 32) {
        __syncthreads();
        // A: W rows, K(c) contiguous -> direct, fp32->bf16 convert
        for (int idx = tid; idx < 128 * 32; idx += 256) {
            int r = idx >> 5, c = idx & 31;
            lA[r * LPITCH + c] = (__bf16)W[(size_t)(m0b + r) * CDIM + k0 + c];
        }
        // B: X is [c][n]; transpose into LDS [n][c], coalesced float2 reads
        for (int idx = tid; idx < 32 * 128; idx += 256) {
            int c = idx >> 7, n2 = (idx & 127) * 2;
            const float2 x2 = *(const float2*)&Xb[(size_t)(k0 + c) * NDIM + n0b + n2];
            lB[(n2    ) * LPITCH + c] = (__bf16)x2.x;
            lB[(n2 + 1) * LPITCH + c] = (__bf16)x2.y;
        }
        __syncthreads();

        v16bf af[4], bfr[4];
#pragma unroll
        for (int mt = 0; mt < 4; ++mt) af[mt]  = load_a_frag(lA, wm + mt * 16, lane);
#pragma unroll
        for (int nt = 0; nt < 4; ++nt) bfr[nt] = load_b_frag(lB, wn + nt * 16, lane);
#pragma unroll
        for (int mt = 0; mt < 4; ++mt)
#pragma unroll
            for (int nt = 0; nt < 4; ++nt)
                acc[mt][nt] = __builtin_amdgcn_wmma_f32_16x16x32_bf16(
                    false, af[mt], false, bfr[nt], (short)0, acc[mt][nt], false, false);
    }

    const int rbase = wm + (lane >> 4) * 8;
    const int cidx  = wn + (lane & 15);
#pragma unroll
    for (int mt = 0; mt < 4; ++mt)
#pragma unroll
        for (int nt = 0; nt < 4; ++nt)
#pragma unroll
            for (int r = 0; r < 8; ++r) {
                const int row = rbase + mt * 16 + r;
                const int col = cidx + nt * 16;
                const float v = acc[mt][nt][r] + lbias[row];
                Yb[(size_t)(m0b + row) * NDIM + n0b + col] = (__bf16)v;
            }
}

// ---------------- Stage 2: S[i][j] = sum_n Q[i][n] K[j][n], tile 128x128, K-step 32
__global__ __launch_bounds__(256) void qk_gemm(
    const __bf16* __restrict__ Q, const __bf16* __restrict__ Km,
    float* __restrict__ S)
{
    __shared__ __bf16 lA[128 * LPITCH];
    __shared__ __bf16 lB[128 * LPITCH];
    const int b   = blockIdx.z;
    const int m0b = blockIdx.y * 128;
    const int n0b = blockIdx.x * 128;
    const __bf16* Qb = Q  + (size_t)b * CDIM * NDIM;
    const __bf16* Kb = Km + (size_t)b * CDIM * NDIM;
    float*        Sb = S  + (size_t)b * CDIM * CDIM;

    const int tid  = threadIdx.x;
    const int lane = tid & 31;
    const int wave = tid >> 5;
    const int wm = (wave & 1) * 64;             // 64x32 per wave
    const int wn = (wave >> 1) * 32;

    v8f acc[4][2] = {};
    for (int k0 = 0; k0 < NDIM; k0 += 32) {
        __syncthreads();
        // Both Q (A) and K (B^T) are already K(n)-contiguous: straight copies
        for (int idx = tid; idx < 128 * 16; idx += 256) {
            int r = idx >> 4, c2 = (idx & 15) * 2;
            *(unsigned int*)&lA[r * LPITCH + c2] =
                *(const unsigned int*)&Qb[(size_t)(m0b + r) * NDIM + k0 + c2];
            *(unsigned int*)&lB[r * LPITCH + c2] =
                *(const unsigned int*)&Kb[(size_t)(n0b + r) * NDIM + k0 + c2];
        }
        __syncthreads();

        v16bf af[4], bfr[2];
#pragma unroll
        for (int mt = 0; mt < 4; ++mt) af[mt]  = load_a_frag(lA, wm + mt * 16, lane);
#pragma unroll
        for (int nt = 0; nt < 2; ++nt) bfr[nt] = load_b_frag(lB, wn + nt * 16, lane);
#pragma unroll
        for (int mt = 0; mt < 4; ++mt)
#pragma unroll
            for (int nt = 0; nt < 2; ++nt)
                acc[mt][nt] = __builtin_amdgcn_wmma_f32_16x16x32_bf16(
                    false, af[mt], false, bfr[nt], (short)0, acc[mt][nt], false, false);
    }

    const int rbase = wm + (lane >> 4) * 8;
    const int cidx  = wn + (lane & 15);
#pragma unroll
    for (int mt = 0; mt < 4; ++mt)
#pragma unroll
        for (int nt = 0; nt < 2; ++nt)
#pragma unroll
            for (int r = 0; r < 8; ++r) {
                const int row = rbase + mt * 16 + r;
                const int col = cidx + nt * 16;
                Sb[(size_t)(m0b + row) * CDIM + n0b + col] = acc[mt][nt][r];
            }
}

// ---------------- Stage 3: row softmax, one wave32 per 512-wide row
__global__ __launch_bounds__(256) void softmax_rows(
    const float* __restrict__ S, __bf16* __restrict__ P)
{
    const int gwave = blockIdx.x * 8 + (threadIdx.x >> 5);  // b*CDIM + i
    const int lane  = threadIdx.x & 31;
    const float* row = S + (size_t)gwave * CDIM;
    __bf16*     prow = P + (size_t)gwave * CDIM;

    float vals[16];
    float m = -3.402823466e38f;
#pragma unroll
    for (int t = 0; t < 16; ++t) { vals[t] = row[lane + 32 * t]; m = fmaxf(m, vals[t]); }
#pragma unroll
    for (int o = 16; o; o >>= 1) m = fmaxf(m, __shfl_xor(m, o, 32));
    float s = 0.f;
#pragma unroll
    for (int t = 0; t < 16; ++t) { vals[t] = __expf(vals[t] - m); s += vals[t]; }
#pragma unroll
    for (int o = 16; o; o >>= 1) s += __shfl_xor(s, o, 32);
    const float inv = 1.f / s;
#pragma unroll
    for (int t = 0; t < 16; ++t) prow[lane + 32 * t] = (__bf16)(vals[t] * inv);
}

// ---------------- Stage 4: out[i][n] = sum_j P[i][j] V[j][n], tile 128x256, K-step 32
__global__ __launch_bounds__(256) void av_gemm(
    const __bf16* __restrict__ P, const __bf16* __restrict__ V,
    float* __restrict__ O)
{
    __shared__ __bf16 lA[128 * LPITCH];
    __shared__ __bf16 lB[256 * LPITCH];
    const int b   = blockIdx.z;
    const int m0b = blockIdx.y * 128;
    const int n0b = blockIdx.x * 256;
    const __bf16* Pb = P + (size_t)b * CDIM * CDIM;
    const __bf16* Vb = V + (size_t)b * CDIM * NDIM;
    float*        Ob = O + (size_t)b * CDIM * NDIM;

    const int tid  = threadIdx.x;
    const int lane = tid & 31;
    const int wave = tid >> 5;
    const int wm = (wave & 1) * 64;
    const int wn = (wave >> 1) * 64;

    v8f acc[4][4] = {};
    for (int k0 = 0; k0 < CDIM; k0 += 32) {
        __syncthreads();
        // A: attn [i][j], j contiguous -> direct copy
        for (int idx = tid; idx < 128 * 16; idx += 256) {
            int r = idx >> 4, c2 = (idx & 15) * 2;
            *(unsigned int*)&lA[r * LPITCH + c2] =
                *(const unsigned int*)&Pb[(size_t)(m0b + r) * CDIM + k0 + c2];
        }
        // B: V is [j][n]; transpose into LDS [n][j], coalesced 32-bit reads
        for (int idx = tid; idx < 32 * 128; idx += 256) {
            int j = idx >> 7, n2 = (idx & 127) * 2;
            const unsigned int u =
                *(const unsigned int*)&Vb[(size_t)(k0 + j) * NDIM + n0b + n2];
            lB[(n2    ) * LPITCH + j] = ((const __bf16*)&u)[0];
            lB[(n2 + 1) * LPITCH + j] = ((const __bf16*)&u)[1];
        }
        __syncthreads();

        v16bf af[4], bfr[4];
#pragma unroll
        for (int mt = 0; mt < 4; ++mt) af[mt]  = load_a_frag(lA, wm + mt * 16, lane);
#pragma unroll
        for (int nt = 0; nt < 4; ++nt) bfr[nt] = load_b_frag(lB, wn + nt * 16, lane);
#pragma unroll
        for (int mt = 0; mt < 4; ++mt)
#pragma unroll
            for (int nt = 0; nt < 4; ++nt)
                acc[mt][nt] = __builtin_amdgcn_wmma_f32_16x16x32_bf16(
                    false, af[mt], false, bfr[nt], (short)0, acc[mt][nt], false, false);
    }

    const int rbase = wm + (lane >> 4) * 8;
    const int cidx  = wn + (lane & 15);
#pragma unroll
    for (int mt = 0; mt < 4; ++mt)
#pragma unroll
        for (int nt = 0; nt < 4; ++nt)
#pragma unroll
            for (int r = 0; r < 8; ++r) {
                const int row = rbase + mt * 16 + r;
                const int col = cidx + nt * 16;
                Ob[(size_t)(m0b + row) * NDIM + n0b + col] = acc[mt][nt][r];
            }
}

extern "C" void kernel_launch(void* const* d_in, const int* in_sizes, int n_in,
                              void* d_out, int out_size, void* d_ws, size_t ws_size,
                              hipStream_t stream) {
    (void)in_sizes; (void)n_in; (void)out_size; (void)ws_size;
    const float* x  = (const float*)d_in[0];
    const float* wq = (const float*)d_in[1];
    const float* bq = (const float*)d_in[2];
    const float* wk = (const float*)d_in[3];
    const float* bk = (const float*)d_in[4];
    const float* wv = (const float*)d_in[5];
    const float* bv = (const float*)d_in[6];
    float* out = (float*)d_out;

    char* ws = (char*)d_ws;
    const size_t szQKV = (size_t)BDIM * CDIM * NDIM * sizeof(__bf16);   // 32 MB each
    __bf16* Q = (__bf16*)(ws);
    __bf16* K = (__bf16*)(ws + szQKV);
    __bf16* V = (__bf16*)(ws + 2 * szQKV);
    float*  S = (float*) (ws + 3 * szQKV);                               // 8 MB
    __bf16* P = (__bf16*)(ws + 3 * szQKV +
                          (size_t)BDIM * CDIM * CDIM * sizeof(float));   // 4 MB

    const dim3 blk(256);
    const dim3 gQKV(NDIM / 256, CDIM / 128, BDIM);   // (16, 4, 8)
    const dim3 gQK (CDIM / 128, CDIM / 128, BDIM);   // (4, 4, 8)
    const dim3 gSM (BDIM * CDIM / 8);                // 512 blocks, wave per row
    const dim3 gAV (NDIM / 256, CDIM / 128, BDIM);   // (16, 4, 8)

    qkv_gemm<<<gQKV, blk, 0, stream>>>(wq, bq, x, Q);
    qkv_gemm<<<gQKV, blk, 0, stream>>>(wk, bk, x, K);
    qkv_gemm<<<gQKV, blk, 0, stream>>>(wv, bv, x, V);
    qk_gemm <<<gQK,  blk, 0, stream>>>(Q, K, S);
    softmax_rows<<<gSM, blk, 0, stream>>>(S, P);
    av_gemm <<<gAV,  blk, 0, stream>>>(P, V, out);
}
